// HunyuanImage3DecoderLayer_82764019794353
// MI455X (gfx1250) — matrix operand order repaired and checked by reading, not compiled, not run on hardware
//
#include <hip/hip_runtime.h>
#include <math.h>

typedef __attribute__((ext_vector_type(16))) __bf16 bf16x16;
typedef __attribute__((ext_vector_type(8)))  float  f32x8;
typedef __attribute__((ext_vector_type(4)))  int    i32x4;
typedef __attribute__((ext_vector_type(4)))  float  f32x4;

union FragB16 { bf16x16 v; i32x4 q[2]; __bf16 e[16]; };

#define S_TOK 2048
#define HIDDEN 2048
#define NHEAD 16
#define NKVH 8
#define HDIM 128
#define NEXP 8
#define CAP 512

static __device__ __forceinline__ f32x8 wmma_bf16(const FragB16& a, const FragB16& b, f32x8 c) {
  return __builtin_amdgcn_wmma_f32_16x16x32_bf16(false, a.v, false, b.v, (short)0, c, false, false);
}

static __device__ __forceinline__ float redmax16(float x) {
  x = fmaxf(x, __shfl_xor(x, 1, 32));
  x = fmaxf(x, __shfl_xor(x, 2, 32));
  x = fmaxf(x, __shfl_xor(x, 4, 32));
  x = fmaxf(x, __shfl_xor(x, 8, 32));
  return x;
}
static __device__ __forceinline__ float redsum16(float x) {
  x += __shfl_xor(x, 1, 32);
  x += __shfl_xor(x, 2, 32);
  x += __shfl_xor(x, 4, 32);
  x += __shfl_xor(x, 8, 32);
  return x;
}

// ---------------------------------------------------------------------------
// RMSNorm: one block (256 thr) per token row of 2048
// ---------------------------------------------------------------------------
__global__ __launch_bounds__(256) void rmsnorm_kernel(const float* __restrict__ x,
                                                      const float* __restrict__ w,
                                                      float* __restrict__ out) {
  const long long t = blockIdx.x;
  const float* xr = x + t * HIDDEN;
  float ss = 0.f;
  for (int i = threadIdx.x; i < HIDDEN; i += 256) { float v = xr[i]; ss += v * v; }
  for (int m = 16; m >= 1; m >>= 1) ss += __shfl_xor(ss, m, 32);
  __shared__ float red[8];
  if ((threadIdx.x & 31) == 0) red[threadIdx.x >> 5] = ss;
  __syncthreads();
  float tot = 0.f;
  #pragma unroll
  for (int i = 0; i < 8; ++i) tot += red[i];
  float inv = rsqrtf(tot / (float)HIDDEN + 1e-5f);
  for (int i = threadIdx.x; i < HIDDEN; i += 256) out[t * HIDDEN + i] = xr[i] * inv * w[i];
}

// ---------------------------------------------------------------------------
// Generic (batched) GEMM: C[M,N] = A[M,K] @ B[K,N] (+Cadd), fp32 in global,
// bf16 in LDS, v_wmma_f32_16x16x32_bf16 compute. Tiles 128x128x32, 8 waves.
// Double-buffered LDS: global loads for tile k+1 overlap WMMA on tile k,
// single barrier per K-step. Assumes M%128==0, N%128==0, K%32==0.
// ---------------------------------------------------------------------------
__global__ __launch_bounds__(256) void gemm_bf16_wmma_kernel(
    const float* __restrict__ A, const float* __restrict__ B,
    float* __restrict__ C, const float* __restrict__ Cadd,
    int M, int N, int K, int lda, int ldb, int ldc,
    long long sA, long long sB, long long sC) {
  (void)M; (void)N;
  __shared__ __bf16 As[2][128 * 40];   // [row][k], pad 40 (80B rows, 16B-aligned)
  __shared__ __bf16 Bs[2][128 * 40];   // [col][k] (transposed on store)
  const int bz = blockIdx.z;
  A += (long long)bz * sA;
  B += (long long)bz * sB;
  C += (long long)bz * sC;
  const int m0 = blockIdx.y * 128;
  const int n0 = blockIdx.x * 128;
  const int tid = threadIdx.x;
  const int lane = tid & 31, wid = tid >> 5;
  const int wm = (wid & 3) * 32, wn = (wid >> 2) * 64;
  const int hl = lane >> 4, ln = lane & 15;

  // per-thread staging coordinates (constant across K-steps)
  const int aRow = tid >> 3, aCol = (tid & 7) * 4;      // + 32*i rows
  const int bRow = tid >> 5, bCol = (tid & 31) * 4;     // + 8*i rows

  auto loadA = [&](int k0, f32x4* reg) {
    #pragma unroll
    for (int i = 0; i < 4; ++i)
      reg[i] = *(const f32x4*)(A + (long long)(m0 + aRow + 32 * i) * lda + k0 + aCol);
  };
  auto loadB = [&](int k0, f32x4* reg) {
    #pragma unroll
    for (int i = 0; i < 4; ++i)
      reg[i] = *(const f32x4*)(B + (long long)(k0 + bRow + 8 * i) * ldb + n0 + bCol);
  };
  auto storeTiles = [&](int buf, const f32x4* ar, const f32x4* br) {
    #pragma unroll
    for (int i = 0; i < 4; ++i) {
      #pragma unroll
      for (int j = 0; j < 4; ++j) {
        As[buf][(aRow + 32 * i) * 40 + aCol + j] = (__bf16)ar[i][j];
        Bs[buf][(bCol + j) * 40 + bRow + 8 * i] = (__bf16)br[i][j];
      }
    }
  };

  const f32x8 fz = {0.f,0.f,0.f,0.f,0.f,0.f,0.f,0.f};
  f32x8 acc[2][4];
  #pragma unroll
  for (int mi = 0; mi < 2; ++mi)
    #pragma unroll
    for (int ni = 0; ni < 4; ++ni) acc[mi][ni] = fz;

  f32x4 aReg[4], bReg[4];
  loadA(0, aReg);
  loadB(0, bReg);
  storeTiles(0, aReg, bReg);
  __syncthreads();

  const int nk = K >> 5;
  for (int kt = 0; kt < nk; ++kt) {
    const int cur = kt & 1;
    const bool more = (kt + 1 < nk);
    if (more) { loadA((kt + 1) * 32, aReg); loadB((kt + 1) * 32, bReg); }

    FragB16 af[2], bf[4];
    #pragma unroll
    for (int mi = 0; mi < 2; ++mi) {
      int row = wm + mi * 16 + ln;
      af[mi].q[0] = *(const i32x4*)&As[cur][row * 40 + 8 * hl];
      af[mi].q[1] = *(const i32x4*)&As[cur][row * 40 + 16 + 8 * hl];
    }
    #pragma unroll
    for (int ni = 0; ni < 4; ++ni) {
      int col = wn + ni * 16 + ln;
      bf[ni].q[0] = *(const i32x4*)&Bs[cur][col * 40 + 8 * hl];
      bf[ni].q[1] = *(const i32x4*)&Bs[cur][col * 40 + 16 + 8 * hl];
    }
    #pragma unroll
    for (int mi = 0; mi < 2; ++mi)
      #pragma unroll
      for (int ni = 0; ni < 4; ++ni)
        acc[mi][ni] = wmma_bf16(af[mi], bf[ni], acc[mi][ni]);

    if (more) storeTiles(cur ^ 1, aReg, bReg);
    __syncthreads();
  }

  // Epilogue: uniform branch hoisted; batched residual loads.
  const long long cbase = (long long)(m0 + wm + 8 * hl) * ldc + (n0 + wn + ln);
  if (Cadd) {
    Cadd += (long long)bz * sC;
    #pragma unroll
    for (int mi = 0; mi < 2; ++mi)
      #pragma unroll
      for (int ni = 0; ni < 4; ++ni) {
        const long long base = cbase + (long long)(mi * 16) * ldc + ni * 16;
        float add[8];
        #pragma unroll
        for (int r = 0; r < 8; ++r) add[r] = Cadd[base + (long long)r * ldc];
        #pragma unroll
        for (int r = 0; r < 8; ++r) C[base + (long long)r * ldc] = acc[mi][ni][r] + add[r];
      }
  } else {
    #pragma unroll
    for (int mi = 0; mi < 2; ++mi)
      #pragma unroll
      for (int ni = 0; ni < 4; ++ni) {
        const long long base = cbase + (long long)(mi * 16) * ldc + ni * 16;
        #pragma unroll
        for (int r = 0; r < 8; ++r) C[base + (long long)r * ldc] = acc[mi][ni][r];
      }
  }
}

// ---------------------------------------------------------------------------
// RoPE in place. grid = S*nh blocks of 128 threads (one head-vector per block).
// ---------------------------------------------------------------------------
__global__ __launch_bounds__(128) void rope_kernel(float* __restrict__ x,
                                                   const float* __restrict__ cs,
                                                   const float* __restrict__ sn,
                                                   int nh) {
  const int tok = blockIdx.x / nh;
  const int d = threadIdx.x;
  float* xr = x + (long long)blockIdx.x * HDIM;
  float c = cs[tok * HDIM + d];
  float s = sn[tok * HDIM + d];
  float v = xr[d];
  float vr = (d < 64) ? -xr[d + 64] : xr[d - 64];
  __syncthreads();
  xr[d] = v * c + vr * s;
}

// ---------------------------------------------------------------------------
// Flash attention, causal, GQA (head h uses kv head h/2).
// grid = (S/128, NH), 256 threads. Each wave owns 16 q rows; block stages
// 32-key K/V tiles in double-buffered LDS. QK^T and P*V are WMMA bf16.
// ---------------------------------------------------------------------------
__global__ __launch_bounds__(256) void attn_kernel(const float* __restrict__ q,
                                                   const float* __restrict__ kk,
                                                   const float* __restrict__ vv,
                                                   float* __restrict__ ctx) {
  __shared__ __bf16 Ks[2][32 * 136];     // [key][hd]
  __shared__ __bf16 Vt[2][128 * 40];     // [hd][key]
  __shared__ __bf16 Ps[8 * 16 * 40];     // per-wave 16x32 P scratch
  const int h = blockIdx.y;
  const int kvh = h >> 1;
  const int qblk = blockIdx.x * 128;
  const int tid = threadIdx.x;
  const int lane = tid & 31, wid = tid >> 5;
  const int hl = lane >> 4, ln = lane & 15;
  const int qt0 = qblk + wid * 16;

  // staging coords
  const int sKey = tid >> 5;             // + 8*i
  const int sC4 = (tid & 31) * 4;

  auto loadKV = [&](int kb, f32x4* kr, f32x4* vr) {
    #pragma unroll
    for (int i = 0; i < 4; ++i) {
      long long tok = (long long)kb * 32 + sKey + 8 * i;
      kr[i] = *(const f32x4*)(kk + tok * (NKVH * HDIM) + kvh * HDIM + sC4);
      vr[i] = *(const f32x4*)(vv + tok * (NKVH * HDIM) + kvh * HDIM + sC4);
    }
  };
  auto storeKV = [&](int buf, const f32x4* kr, const f32x4* vr) {
    #pragma unroll
    for (int i = 0; i < 4; ++i)
      #pragma unroll
      for (int j = 0; j < 4; ++j) {
        Ks[buf][(sKey + 8 * i) * 136 + sC4 + j] = (__bf16)kr[i][j];
        Vt[buf][(sC4 + j) * 40 + sKey + 8 * i] = (__bf16)vr[i][j];
      }
  };

  // Q fragments (16x128 as 4 A-frags of 16x32) direct from global
  FragB16 qf[4];
  {
    const float* qrow = q + (long long)(qt0 + ln) * (NHEAD * HDIM) + h * HDIM;
    #pragma unroll
    for (int c = 0; c < 4; ++c)
      #pragma unroll
      for (int e2 = 0; e2 < 16; ++e2) {
        int kidx = c * 32 + (e2 & 7) + 8 * hl + 16 * (e2 >> 3);
        qf[c].e[e2] = (__bf16)qrow[kidx];
      }
  }

  const f32x8 fz = {0.f,0.f,0.f,0.f,0.f,0.f,0.f,0.f};
  f32x8 o[8];
  #pragma unroll
  for (int j = 0; j < 8; ++j) o[j] = fz;
  float mr[8], lr[8];
  #pragma unroll
  for (int r = 0; r < 8; ++r) { mr[r] = -1e30f; lr[r] = 0.f; }

  const int kbmax = (qblk + 127) >> 5;
  f32x4 kReg[4], vReg[4];
  loadKV(0, kReg, vReg);
  storeKV(0, kReg, vReg);
  __syncthreads();

  for (int kb = 0; kb <= kbmax; ++kb) {
    const int cur = kb & 1;
    const bool more = (kb < kbmax);
    if (more) loadKV(kb + 1, kReg, vReg);

    if (kb * 32 <= qt0 + 15) {          // wave-uniform predicate: EXEC stays full
      f32x8 s[2];
      #pragma unroll
      for (int nt = 0; nt < 2; ++nt) {
        s[nt] = fz;
        #pragma unroll
        for (int c = 0; c < 4; ++c) {
          FragB16 kf;
          int key = nt * 16 + ln;
          kf.q[0] = *(const i32x4*)&Ks[cur][key * 136 + c * 32 + 8 * hl];
          kf.q[1] = *(const i32x4*)&Ks[cur][key * 136 + c * 32 + 16 + 8 * hl];
          s[nt] = wmma_bf16(qf[c], kf, s[nt]);
        }
      }
      const int key0 = kb * 32 + ln;
      #pragma unroll
      for (int r = 0; r < 8; ++r) {
        int rowg = qt0 + r + 8 * hl;
        float s0 = s[0][r] * 0.088388347648318447f;
        float s1 = s[1][r] * 0.088388347648318447f;
        if (key0 > rowg)      s0 = -1e30f;
        if (key0 + 16 > rowg) s1 = -1e30f;
        float tm = fmaxf(redmax16(s0), redmax16(s1));
        float nm = fmaxf(mr[r], tm);
        float fac = __expf(mr[r] - nm);
        float p0 = __expf(s0 - nm);
        float p1 = __expf(s1 - nm);
        float rs = redsum16(p0) + redsum16(p1);
        lr[r] = lr[r] * fac + rs;
        mr[r] = nm;
        #pragma unroll
        for (int j = 0; j < 8; ++j) o[j][r] *= fac;
        Ps[wid * 640 + (r + 8 * hl) * 40 + ln] = (__bf16)p0;
        Ps[wid * 640 + (r + 8 * hl) * 40 + 16 + ln] = (__bf16)p1;
      }
      FragB16 pf;
      pf.q[0] = *(const i32x4*)&Ps[wid * 640 + ln * 40 + 8 * hl];
      pf.q[1] = *(const i32x4*)&Ps[wid * 640 + ln * 40 + 16 + 8 * hl];
      #pragma unroll
      for (int j = 0; j < 8; ++j) {
        FragB16 vf;
        vf.q[0] = *(const i32x4*)&Vt[cur][(j * 16 + ln) * 40 + 8 * hl];
        vf.q[1] = *(const i32x4*)&Vt[cur][(j * 16 + ln) * 40 + 16 + 8 * hl];
        o[j] = wmma_bf16(pf, vf, o[j]);
      }
    }

    if (more) storeKV(cur ^ 1, kReg, vReg);
    __syncthreads();
  }

  #pragma unroll
  for (int j = 0; j < 8; ++j)
    #pragma unroll
    for (int r = 0; r < 8; ++r) {
      int rowg = qt0 + r + 8 * hl;
      ctx[(long long)rowg * (NHEAD * HDIM) + h * HDIM + j * 16 + ln] = o[j][r] / lr[r];
    }
}

// ---------------------------------------------------------------------------
// SwiGLU: out[m][i] = gu[m][i] * silu(gu[m][I+i])
// ---------------------------------------------------------------------------
__global__ __launch_bounds__(256) void swiglu_kernel(const float* __restrict__ gu,
                                                     float* __restrict__ out, int I) {
  long long m = blockIdx.x;
  const float* g = gu + m * 2 * I;
  float* o = out + m * I;
  for (int i = threadIdx.x; i < I; i += 256) {
    float a = g[i], b = g[I + i];
    o[i] = a * (b / (1.f + __expf(-b)));
  }
}

// ---------------------------------------------------------------------------
// Router gate: one wave per token. logits -> softmax -> top2 -> renorm weights
// ---------------------------------------------------------------------------
__global__ __launch_bounds__(256) void gate_kernel(const float* __restrict__ x,
                                                   const float* __restrict__ gw,
                                                   int* __restrict__ ei,
                                                   float* __restrict__ wts) {
  const int tok = blockIdx.x * 8 + (threadIdx.x >> 5);
  const int lane = threadIdx.x & 31;
  const float* xr = x + (long long)tok * HIDDEN;
  float acc[NEXP];
  #pragma unroll
  for (int e = 0; e < NEXP; ++e) acc[e] = 0.f;
  for (int i = lane; i < HIDDEN; i += 32) {
    float xv = xr[i];
    const float* g = gw + (long long)i * NEXP;
    #pragma unroll
    for (int e = 0; e < NEXP; ++e) acc[e] += xv * g[e];
  }
  #pragma unroll
  for (int e = 0; e < NEXP; ++e)
    for (int m = 16; m >= 1; m >>= 1) acc[e] += __shfl_xor(acc[e], m, 32);
  if (lane == 0) {
    float mx = acc[0];
    #pragma unroll
    for (int e = 1; e < NEXP; ++e) mx = fmaxf(mx, acc[e]);
    float ex[NEXP], ssum = 0.f;
    #pragma unroll
    for (int e = 0; e < NEXP; ++e) { ex[e] = __expf(acc[e] - mx); ssum += ex[e]; }
    int i0 = 0;
    #pragma unroll
    for (int e = 1; e < NEXP; ++e) if (ex[e] > ex[i0]) i0 = e;
    int i1 = (i0 == 0) ? 1 : 0;
    #pragma unroll
    for (int e = 0; e < NEXP; ++e) if (e != i0 && ex[e] > ex[i1]) i1 = e;
    float g0 = ex[i0] / ssum, g1 = ex[i1] / ssum;
    float den = g0 + g1;
    if (den < 1.1920929e-07f) den = 1.1920929e-07f;
    ei[tok * 2 + 0] = i0;  ei[tok * 2 + 1] = i1;
    wts[tok * 2 + 0] = g0 / den;  wts[tok * 2 + 1] = g1 / den;
  }
}

// ---------------------------------------------------------------------------
// Slot assignment: k-major cumsum priority per expert, capacity drop.
// One block of 8 waves; wave w scans all 2T entries for expert w with
// ballot-based running prefix. Deterministic.
// ---------------------------------------------------------------------------
__global__ __launch_bounds__(256) void slot_kernel(const int* __restrict__ ei,
                                                   int* __restrict__ slotb) {
  const int wid = threadIdx.x >> 5;
  const int lane = threadIdx.x & 31;
  int base = 0;
  for (int f0 = 0; f0 < 2 * S_TOK; f0 += 32) {
    int f = f0 + lane;
    int kkk = f >> 11;          // /S_TOK (k-major flatten)
    int t = f & (S_TOK - 1);
    int e = ei[t * 2 + kkk];
    bool match = (e == wid);
    unsigned mask = (unsigned)__ballot(match);
    int pos = base + __popc(mask & ((1u << lane) - 1u));
    if (match) slotb[t * 2 + kkk] = (pos < CAP) ? pos : -1;
    base += __popc(mask);
  }
}

// ---------------------------------------------------------------------------
// Dispatch: copy token row into expert capacity buffer
// ---------------------------------------------------------------------------
__global__ __launch_bounds__(256) void dispatch_kernel(const float* __restrict__ x,
                                                       const int* __restrict__ ei,
                                                       const int* __restrict__ slotb,
                                                       float* __restrict__ disp) {
  const int idx = blockIdx.x;           // t*2 + k
  const int slot = slotb[idx];
  if (slot < 0) return;
  const int t = idx >> 1;
  const int e = ei[idx];
  const float* src = x + (long long)t * HIDDEN;
  float* dst = disp + ((long long)e * CAP + slot) * HIDDEN;
  for (int i = threadIdx.x; i < HIDDEN; i += 256) dst[i] = src[i];
}

// ---------------------------------------------------------------------------
// Combine: out[t] += sum_k w[t,k] * eo[expert_slot(t,k)]  (skip dropped)
// ---------------------------------------------------------------------------
__global__ __launch_bounds__(256) void combine_kernel(float* __restrict__ out,
                                                      const float* __restrict__ eo,
                                                      const int* __restrict__ ei,
                                                      const int* __restrict__ slotb,
                                                      const float* __restrict__ wts) {
  const int t = blockIdx.x;
  const int s0 = slotb[t * 2 + 0], s1 = slotb[t * 2 + 1];
  const long long r0 = (s0 >= 0) ? ((long long)ei[t * 2 + 0] * CAP + s0) * HIDDEN : 0;
  const long long r1 = (s1 >= 0) ? ((long long)ei[t * 2 + 1] * CAP + s1) * HIDDEN : 0;
  const float w0 = wts[t * 2 + 0], w1 = wts[t * 2 + 1];
  for (int i = threadIdx.x; i < HIDDEN; i += 256) {
    float a = out[(long long)t * HIDDEN + i];
    if (s0 >= 0) a += w0 * eo[r0 + i];
    if (s1 >= 0) a += w1 * eo[r1 + i];
    out[(long long)t * HIDDEN + i] = a;
  }
}

// ---------------------------------------------------------------------------
extern "C" void kernel_launch(void* const* d_in, const int* in_sizes, int n_in,
                              void* d_out, int out_size, void* d_ws, size_t ws_size,
                              hipStream_t stream) {
  (void)in_sizes; (void)n_in; (void)out_size; (void)ws_size;
  const float* h    = (const float*)d_in[0];
  const float* cosb = (const float*)d_in[1];
  const float* sinb = (const float*)d_in[2];
  const float* wq   = (const float*)d_in[3];
  const float* wk   = (const float*)d_in[4];
  const float* wv   = (const float*)d_in[5];
  const float* wo   = (const float*)d_in[6];
  const float* ln1  = (const float*)d_in[7];
  const float* ln2  = (const float*)d_in[8];
  const float* gw   = (const float*)d_in[9];
  const float* sguw = (const float*)d_in[10];
  const float* sdw  = (const float*)d_in[11];
  const float* eguw = (const float*)d_in[12];
  const float* edw  = (const float*)d_in[13];
  float* out = (float*)d_out;

  float* ws = (float*)d_ws;
  size_t off = 0;
  auto alloc = [&](size_t n) { float* p = ws + off; off += n; return p; };
  float* xn1  = alloc(4194304);              // [T, H]
  float* qb   = alloc(4194304);              // [T, NH*HD]
  float* kb   = alloc(2097152);              // [T, NKV*HD]
  float* vb   = alloc(2097152);
  float* ctx  = alloc(4194304);
  float* h2   = alloc(4194304);
  float* xn2  = alloc(4194304);
  float* sgu  = alloc(4194304);              // [T, 2*I_SH]
  float* sswi = alloc(2097152);              // [T, I_SH]
  float* disp = alloc((size_t)NEXP * CAP * HIDDEN);       // 8M
  float* egu  = alloc((size_t)NEXP * CAP * 2048);         // 8M
  float* eswi = alloc((size_t)NEXP * CAP * 1024);         // 4M
  float* eo   = alloc((size_t)NEXP * CAP * HIDDEN);       // 8M
  float* wts  = alloc(2 * S_TOK);
  int*   ei    = (int*)alloc(2 * S_TOK);
  int*   slotb = (int*)alloc(2 * S_TOK);

  // --- attention block ---
  rmsnorm_kernel<<<S_TOK, 256, 0, stream>>>(h, ln1, xn1);
  gemm_bf16_wmma_kernel<<<dim3(16,16,1),256,0,stream>>>(xn1, wq, qb, nullptr,
      2048,2048,2048, 2048,2048,2048, 0,0,0);
  gemm_bf16_wmma_kernel<<<dim3(8,16,1),256,0,stream>>>(xn1, wk, kb, nullptr,
      2048,1024,2048, 2048,1024,1024, 0,0,0);
  gemm_bf16_wmma_kernel<<<dim3(8,16,1),256,0,stream>>>(xn1, wv, vb, nullptr,
      2048,1024,2048, 2048,1024,1024, 0,0,0);
  rope_kernel<<<S_TOK * NHEAD, 128, 0, stream>>>(qb, cosb, sinb, NHEAD);
  rope_kernel<<<S_TOK * NKVH, 128, 0, stream>>>(kb, cosb, sinb, NKVH);
  attn_kernel<<<dim3(S_TOK/128, NHEAD), 256, 0, stream>>>(qb, kb, vb, ctx);
  gemm_bf16_wmma_kernel<<<dim3(16,16,1),256,0,stream>>>(ctx, wo, h2, h,   // + residual
      2048,2048,2048, 2048,2048,2048, 0,0,0);

  // --- MLP block ---
  rmsnorm_kernel<<<S_TOK, 256, 0, stream>>>(h2, ln2, xn2);
  gemm_bf16_wmma_kernel<<<dim3(16,16,1),256,0,stream>>>(xn2, sguw, sgu, nullptr,
      2048,2048,2048, 2048,2048,2048, 0,0,0);
  swiglu_kernel<<<S_TOK, 256, 0, stream>>>(sgu, sswi, 1024);
  gemm_bf16_wmma_kernel<<<dim3(16,16,1),256,0,stream>>>(sswi, sdw, out, h2, // out = h2 + shared
      2048,2048,1024, 1024,2048,2048, 0,0,0);

  // --- MoE routed ---
  gate_kernel<<<S_TOK/8, 256, 0, stream>>>(xn2, gw, ei, wts);
  slot_kernel<<<1, 256, 0, stream>>>(ei, slotb);
  dispatch_kernel<<<2 * S_TOK, 256, 0, stream>>>(xn2, ei, slotb, disp);
  gemm_bf16_wmma_kernel<<<dim3(16,4,NEXP),256,0,stream>>>(disp, eguw, egu, nullptr,
      512,2048,2048, 2048,2048,2048, 512LL*2048, 2048LL*2048, 512LL*2048);
  swiglu_kernel<<<NEXP * CAP, 256, 0, stream>>>(egu, eswi, 1024);
  gemm_bf16_wmma_kernel<<<dim3(16,4,NEXP),256,0,stream>>>(eswi, edw, eo, nullptr,
      512,2048,1024, 1024,2048,2048, 512LL*1024, 1024LL*2048, 512LL*2048);
  combine_kernel<<<S_TOK, 256, 0, stream>>>(out, eo, ei, slotb, wts);
}
